// FluxDoubleBlock_3204045603062
// MI455X (gfx1250) — matrix operand order repaired and checked by reading, not compiled, not run on hardware
//
#include <hip/hip_runtime.h>
#include <cstddef>

// ---------------- problem constants (Flux double block) ----------------
#define LI   1024
#define LT   512
#define LL   1536          // joint length
#define DD   3072
#define NH   24
#define HD   128
#define M3   9216          // 3*D
#define MMH  12288         // MLP hidden M
#define GUW  24576         // 2*M
#define MODN 18432         // 6*D
#define EPSF 1e-6f

// ---------------- vector types for WMMA ----------------
typedef __attribute__((ext_vector_type(16))) __bf16          v16bf;
typedef __attribute__((ext_vector_type(8)))  float           v8f;
typedef __attribute__((ext_vector_type(16))) unsigned short  us16;
typedef __attribute__((ext_vector_type(8)))  unsigned short  us8;

__device__ __forceinline__ unsigned short f2bf(float f) {
  unsigned int u = __float_as_uint(f);
  u += 0x7FFFu + ((u >> 16) & 1u);          // round-to-nearest-even
  return (unsigned short)(u >> 16);
}
// packed pair convert: use HW v_cvt_pk_bf16_f32 when the builtin exists
__device__ __forceinline__ unsigned int f2bf2(float a, float b) {
#if __has_builtin(__builtin_amdgcn_cvt_pk_bf16_f32)
  auto r = __builtin_amdgcn_cvt_pk_bf16_f32(a, b);
  unsigned int u;
  __builtin_memcpy(&u, &r, 4);
  return u;
#else
  return (unsigned int)f2bf(a) | ((unsigned int)f2bf(b) << 16);
#endif
}
__device__ __forceinline__ float silu_f(float x) {
  return x / (1.0f + __expf(-x));
}
// assemble a 16-element bf16 fragment from two 16B-aligned 8-element chunks
__device__ __forceinline__ v16bf frag16(const unsigned short* p0,
                                        const unsigned short* p1) {
  us8 lo = *(const us8*)p0;
  us8 hi = *(const us8*)p1;
  us16 u = __builtin_shufflevector(lo, hi, 0,1,2,3,4,5,6,7,8,9,10,11,12,13,14,15);
  return __builtin_bit_cast(v16bf, u);
}

// ======================= small elementwise kernels =======================
__global__ void silu_vec_k(const float* __restrict__ x, float* __restrict__ y, int n) {
  int i = blockIdx.x * 256 + threadIdx.x;
  if (i < n) y[i] = silu_f(x[i]);
}

// out[j] = bias[j] + sum_k t[k]*W[k*N+j]   (N = 18432, K = 3072)
__global__ void modvec_k(const float* __restrict__ t, const float* __restrict__ W,
                         const float* __restrict__ bias, float* __restrict__ out) {
  __shared__ float tsh[DD];
  for (int i = threadIdx.x; i < DD; i += 256) tsh[i] = t[i];
  __syncthreads();
  int j = blockIdx.x * 256 + threadIdx.x;
  float acc = bias[j];
  for (int k = 0; k < DD; ++k) acc += tsh[k] * W[(size_t)k * MODN + j];
  out[j] = acc;
}

// convert W[K][N] fp32 -> Wt[N][K] bf16 (transpose through LDS, 64x64 tiles)
__global__ __launch_bounds__(256)
void convtr_k(const float* __restrict__ W, unsigned short* __restrict__ Wt,
              int Kdim, int Ndim) {
  __shared__ unsigned short t[64][72];
  int k0 = blockIdx.y * 64, n0 = blockIdx.x * 64;
  int tid = threadIdx.x;
  {
    int r = tid >> 2;            // k row in tile (0..63)
    int c = (tid & 3) * 16;      // n col group
    const float* src = W + (size_t)(k0 + r) * Ndim + n0 + c;
#pragma unroll
    for (int i = 0; i < 16; i += 2) {
      unsigned int p = f2bf2(src[i], src[i + 1]);
      t[c + i][r]     = (unsigned short)p;
      t[c + i + 1][r] = (unsigned short)(p >> 16);
    }
  }
  __syncthreads();
  {
    int r = tid >> 2;            // n row in tile
    int c = (tid & 3) * 16;      // k col group
    unsigned short* dst = Wt + (size_t)(n0 + r) * Kdim + k0 + c;
#pragma unroll
    for (int i = 0; i < 16; ++i) dst[i] = t[r][c + i];
  }
}

// h1[row] = rms(x)*w * (1+scale1) + shift1 -> bf16 ; joint order: txt rows then img
__global__ void norm_mod1_k(const float* __restrict__ img, const float* __restrict__ txt,
                            const float* __restrict__ img_w, const float* __restrict__ txt_w,
                            const float* __restrict__ mimg, const float* __restrict__ mtxt,
                            unsigned short* __restrict__ h) {
  int row = blockIdx.x;
  const float *x, *w, *mod;
  if (row < LT) { x = txt + (size_t)row * DD;        w = txt_w; mod = mtxt; }
  else          { x = img + (size_t)(row - LT) * DD; w = img_w; mod = mimg; }
  __shared__ float red[256];
  float s = 0.f;
  for (int c = threadIdx.x; c < DD; c += 256) { float v = x[c]; s += v * v; }
  red[threadIdx.x] = s; __syncthreads();
  for (int st = 128; st > 0; st >>= 1) {
    if (threadIdx.x < st) red[threadIdx.x] += red[threadIdx.x + st];
    __syncthreads();
  }
  float r = rsqrtf(red[0] / (float)DD + EPSF);
  unsigned short* y = h + (size_t)row * DD;
  for (int c = threadIdx.x; c < DD; c += 256)
    y[c] = f2bf(x[c] * r * w[c] * (1.0f + mod[DD + c]) + mod[c]);
}

// h2[row] = rms(cur)*w*(1+scale2)+shift2 -> bf16 ; rows: img then txt (reads d_out)
__global__ void norm_mod2_k(const float* __restrict__ cur,
                            const float* __restrict__ img_w, const float* __restrict__ txt_w,
                            const float* __restrict__ mimg, const float* __restrict__ mtxt,
                            unsigned short* __restrict__ h) {
  int row = blockIdx.x;
  const float* x = cur + (size_t)row * DD;
  const float *w, *mod;
  if (row < LI) { w = img_w; mod = mimg; } else { w = txt_w; mod = mtxt; }
  __shared__ float red[256];
  float s = 0.f;
  for (int c = threadIdx.x; c < DD; c += 256) { float v = x[c]; s += v * v; }
  red[threadIdx.x] = s; __syncthreads();
  for (int st = 128; st > 0; st >>= 1) {
    if (threadIdx.x < st) red[threadIdx.x] += red[threadIdx.x + st];
    __syncthreads();
  }
  float r = rsqrtf(red[0] / (float)DD + EPSF);
  unsigned short* y = h + (size_t)row * DD;
  for (int c = threadIdx.x; c < DD; c += 256)
    y[c] = f2bf(x[c] * r * w[c] * (1.0f + mod[4 * DD + c]) + mod[3 * DD + c]);
}

// d_out = residual + gate1 * proj(attn) ; joint rows: txt first, out rows: img first
__global__ void resid_attn_k(const float* __restrict__ img, const float* __restrict__ txt,
                             const float* __restrict__ proj,
                             const float* __restrict__ mimg, const float* __restrict__ mtxt,
                             float* __restrict__ out) {
  int idx = blockIdx.x * 256 + threadIdx.x;    // over LL*DD
  int row = idx / DD, c = idx - row * DD;
  if (row < LI)
    out[idx] = img[(size_t)row * DD + c] + mimg[2 * DD + c] * proj[(size_t)(LT + row) * DD + c];
  else {
    int tr = row - LI;
    out[idx] = txt[(size_t)tr * DD + c] + mtxt[2 * DD + c] * proj[(size_t)tr * DD + c];
  }
}

// act = silu(gate)*up -> bf16 ; gu rows have [gate | up]
__global__ void silu_gate_k(const float* __restrict__ gu, unsigned short* __restrict__ act,
                            int total) {
  int idx = blockIdx.x * 256 + threadIdx.x;    // total = rows*MMH
  if (idx >= total) return;
  int i = idx / MMH, j = idx - i * MMH;
  float g = gu[(size_t)i * GUW + j];
  float u = gu[(size_t)i * GUW + MMH + j];
  act[idx] = f2bf(silu_f(g) * u);
}

// outp += gate2 * mlp
__global__ void resid_mlp_k(const float* __restrict__ mlp, const float* __restrict__ mod,
                            float* __restrict__ outp, int total) {
  int idx = blockIdx.x * 256 + threadIdx.x;
  if (idx >= total) return;
  int c = idx % DD;
  outp[idx] += mod[5 * DD + c] * mlp[idx];
}

// =============== q/k RMS-norm + bf16 pack (one wave per token,head) ===============
__global__ __launch_bounds__(32)
void qk_norm_pack_k(const float* __restrict__ qkv,
                    const float* __restrict__ qw, const float* __restrict__ kw,
                    unsigned short* __restrict__ Q, unsigned short* __restrict__ K,
                    unsigned short* __restrict__ V) {
  int token = blockIdx.x, h = blockIdx.y, lane = threadIdx.x;
  const float* qr = qkv + (size_t)token * M3 + h * HD;
  const float* kr = qr + DD;
  const float* vr = qr + 2 * DD;
  float q[4], k[4], sq = 0.f, sk = 0.f;
#pragma unroll
  for (int i = 0; i < 4; ++i) {
    int c = lane + 32 * i;
    q[i] = qr[c]; k[i] = kr[c];
    sq += q[i] * q[i]; sk += k[i] * k[i];
  }
#pragma unroll
  for (int m = 16; m >= 1; m >>= 1) { sq += __shfl_xor(sq, m); sk += __shfl_xor(sk, m); }
  float rq = rsqrtf(sq / (float)HD + EPSF);
  float rk = rsqrtf(sk / (float)HD + EPSF);
  size_t o = (size_t)token * DD + h * HD;
#pragma unroll
  for (int i = 0; i < 4; ++i) {
    int c = lane + 32 * i;
    Q[o + c] = f2bf(q[i] * rq * qw[c]);
    K[o + c] = f2bf(k[i] * rk * kw[c]);
    V[o + c] = f2bf(vr[c]);
  }
}

// ======================= bf16 WMMA GEMM (register streaming, no LDS) ===========
// C[M,N] = A[M,K] @ B[K,N] (+bias).  A: bf16 [M][K].  Bt: bf16 [N][K] (pre-transposed).
// 256 threads = 8 waves; block tile 128x128; wave tile 32x64 (2x4 wmma tiles); BK=32.
// Fragments are contiguous global loads, same k order in A and B (k = 16*(lane/16) + s); reuse across
// waves/blocks is served by WGP$/L2 (bf16 weights fit the 192MB L2).
// K-loop unrolled x2 with ping-pong fragment buffers: loads for step k+32 stay in
// flight under step k's WMMAs, and no register moves are needed to rotate buffers.
// Requires Kdim % 64 == 0 (true for all K here: 3072, 12288).
#define LOAD_SET(A0, A1, B0, B1, B2, B3, kk)                        \
  A0 = frag16(ap0 + (kk) + g16, ap0 + (kk) + g16 + 8);              \
  A1 = frag16(ap1 + (kk) + g16, ap1 + (kk) + g16 + 8);              \
  B0 = frag16(bp0 + (kk), bp0 + (kk) + 8);                          \
  B1 = frag16(bp1 + (kk), bp1 + (kk) + 8);                          \
  B2 = frag16(bp2 + (kk), bp2 + (kk) + 8);                          \
  B3 = frag16(bp3 + (kk), bp3 + (kk) + 8);

#define WMMA_SET(A0, A1, B0, B1, B2, B3)                                             \
  acc[0][0] = __builtin_amdgcn_wmma_f32_16x16x32_bf16(false, A0, false, B0, (short)0,\
                                                      acc[0][0], false, false);      \
  acc[1][0] = __builtin_amdgcn_wmma_f32_16x16x32_bf16(false, A1, false, B0, (short)0,\
                                                      acc[1][0], false, false);      \
  acc[0][1] = __builtin_amdgcn_wmma_f32_16x16x32_bf16(false, A0, false, B1, (short)0,\
                                                      acc[0][1], false, false);      \
  acc[1][1] = __builtin_amdgcn_wmma_f32_16x16x32_bf16(false, A1, false, B1, (short)0,\
                                                      acc[1][1], false, false);      \
  acc[0][2] = __builtin_amdgcn_wmma_f32_16x16x32_bf16(false, A0, false, B2, (short)0,\
                                                      acc[0][2], false, false);      \
  acc[1][2] = __builtin_amdgcn_wmma_f32_16x16x32_bf16(false, A1, false, B2, (short)0,\
                                                      acc[1][2], false, false);      \
  acc[0][3] = __builtin_amdgcn_wmma_f32_16x16x32_bf16(false, A0, false, B3, (short)0,\
                                                      acc[0][3], false, false);      \
  acc[1][3] = __builtin_amdgcn_wmma_f32_16x16x32_bf16(false, A1, false, B3, (short)0,\
                                                      acc[1][3], false, false);

__global__ __launch_bounds__(256)
void gemm_bf16_k(const unsigned short* __restrict__ A, const unsigned short* __restrict__ Bt,
                 const float* __restrict__ bias, float* __restrict__ C,
                 int Ndim, int Kdim) {
  int tid = threadIdx.x;
  int wid = tid >> 5, lane = tid & 31;
  int lm = lane & 15, g = lane >> 4;
  int bm0 = blockIdx.y * 128, bn0 = blockIdx.x * 128;
  int wm = (wid >> 1) * 32;        // 0,32,64,96
  int wn = (wid & 1) * 64;         // 0,64
  int g8 = g * 8, g16 = g * 16;

  const unsigned short* ap0 = A + (size_t)(bm0 + wm + lm) * Kdim;
  const unsigned short* ap1 = A + (size_t)(bm0 + wm + 16 + lm) * Kdim;
  const unsigned short* bp0 = Bt + (size_t)(bn0 + wn + lm) * Kdim + g16;
  const unsigned short* bp1 = Bt + (size_t)(bn0 + wn + 16 + lm) * Kdim + g16;
  const unsigned short* bp2 = Bt + (size_t)(bn0 + wn + 32 + lm) * Kdim + g16;
  const unsigned short* bp3 = Bt + (size_t)(bn0 + wn + 48 + lm) * Kdim + g16;

  v8f acc[2][4];
#pragma unroll
  for (int i = 0; i < 2; ++i)
#pragma unroll
    for (int j = 0; j < 4; ++j) acc[i][j] = (v8f){0.f,0.f,0.f,0.f,0.f,0.f,0.f,0.f};

  v16bf a0c, a1c, bc0, bc1, bc2, bc3;   // "current" fragment set
  v16bf a0n, a1n, bn0_, bn1_, bn2_, bn3_; // "next" fragment set

  LOAD_SET(a0c, a1c, bc0, bc1, bc2, bc3, 0)

  int k = 0;
  for (; k + 64 < Kdim; k += 64) {
    LOAD_SET(a0n, a1n, bn0_, bn1_, bn2_, bn3_, k + 32)
    __builtin_prefetch(ap0 + k + 288, 0, 1);
    __builtin_prefetch(ap1 + k + 288, 0, 1);
    WMMA_SET(a0c, a1c, bc0, bc1, bc2, bc3)
    LOAD_SET(a0c, a1c, bc0, bc1, bc2, bc3, k + 64)
    WMMA_SET(a0n, a1n, bn0_, bn1_, bn2_, bn3_)
  }
  // tail: steps k and k+32 (Kdim % 64 == 0 guarantees exactly two remain)
  LOAD_SET(a0n, a1n, bn0_, bn1_, bn2_, bn3_, k + 32)
  WMMA_SET(a0c, a1c, bc0, bc1, bc2, bc3)
  WMMA_SET(a0n, a1n, bn0_, bn1_, bn2_, bn3_)

#pragma unroll
  for (int i = 0; i < 2; ++i)
#pragma unroll
    for (int j = 0; j < 4; ++j) {
      int col = bn0 + wn + j * 16 + lm;
      float bv = bias ? bias[col] : 0.f;
#pragma unroll
      for (int r = 0; r < 8; ++r) {
        int row = bm0 + wm + i * 16 + g * 8 + r;
        C[(size_t)row * Ndim + col] = acc[i][j][r] + bv;
      }
    }
}

// ======================= flash attention (bf16 WMMA) =======================
// grid: (L/128 q-tiles, NH heads), 256 threads = 8 waves, each wave = 16 q rows.
#define BKP 40
__global__ __launch_bounds__(256)
void attn_flash_k(const unsigned short* __restrict__ Q, const unsigned short* __restrict__ K,
                  const unsigned short* __restrict__ V, unsigned short* __restrict__ O) {
  __shared__ unsigned short vt[HD][BKP];        // V^T tile: [hd][key 0..31]
  __shared__ unsigned short pb[8][16][BKP];     // per-wave P tile 16x32
  int tid = threadIdx.x, wid = tid >> 5, lane = tid & 31;
  int lm = lane & 15, g = lane >> 4;
  int h = blockIdx.y;
  int qrow = blockIdx.x * 128 + wid * 16;       // wave's first q row
  const float scale = 0.08838834764831845f;     // 1/sqrt(128)

  float mrow[8], lrow[8];
#pragma unroll
  for (int r = 0; r < 8; ++r) { mrow[r] = -1e30f; lrow[r] = 0.f; }
  v8f o[8];
#pragma unroll
  for (int n = 0; n < 8; ++n) o[n] = (v8f){0.f,0.f,0.f,0.f,0.f,0.f,0.f,0.f};

  const unsigned short* qp_base = Q + (size_t)(qrow + lm) * DD + h * HD;

  for (int kt = 0; kt < LL; kt += 32) {
    __syncthreads();
    { // stage V^T for this key tile (block cooperative)
      int e = tid * 16, key = e >> 7, hd = e & 127;
      const unsigned short* src = V + (size_t)(kt + key) * DD + h * HD + hd;
#pragma unroll
      for (int i = 0; i < 16; ++i) vt[hd + i][key] = src[i];
    }
    __syncthreads();

    // S = Q K^T : two 16x16 tiles, K-dim = HD in 4 steps of 32
    v8f s0 = (v8f){0.f,0.f,0.f,0.f,0.f,0.f,0.f,0.f};
    v8f s1 = s0;
#pragma unroll
    for (int s4 = 0; s4 < 4; ++s4) {
      const unsigned short* qp = qp_base + s4 * 32;
      v16bf a = frag16(qp + g * 16, qp + g * 16 + 8);
      const unsigned short* k0p = K + (size_t)(kt + lm) * DD + h * HD + s4 * 32 + g * 16;
      const unsigned short* k1p = K + (size_t)(kt + 16 + lm) * DD + h * HD + s4 * 32 + g * 16;
      v16bf b0 = frag16(k0p, k0p + 8);
      v16bf b1 = frag16(k1p, k1p + 8);
      s0 = __builtin_amdgcn_wmma_f32_16x16x32_bf16(false, a, false, b0, (short)0, s0, false, false);
      s1 = __builtin_amdgcn_wmma_f32_16x16x32_bf16(false, a, false, b1, (short)0, s1, false, false);
    }

    // online softmax per row (rows g*8+r live across 16 lanes of a half-wave)
    float alpha[8];
#pragma unroll
    for (int r = 0; r < 8; ++r) {
      float v0 = s0[r] * scale, v1 = s1[r] * scale;
      float mx = fmaxf(v0, v1);
#pragma unroll
      for (int m = 8; m >= 1; m >>= 1) mx = fmaxf(mx, __shfl_xor(mx, m));
      float nm = fmaxf(mrow[r], mx);
      alpha[r] = __expf(mrow[r] - nm);
      mrow[r] = nm;
      float p0 = __expf(v0 - nm), p1 = __expf(v1 - nm);
      float ps = p0 + p1;
#pragma unroll
      for (int m = 8; m >= 1; m >>= 1) ps += __shfl_xor(ps, m);
      lrow[r] = lrow[r] * alpha[r] + ps;
      pb[wid][g * 8 + r][lm]      = f2bf(p0);
      pb[wid][g * 8 + r][16 + lm] = f2bf(p1);
    }
#pragma unroll
    for (int n = 0; n < 8; ++n)
#pragma unroll
      for (int r = 0; r < 8; ++r) o[n][r] *= alpha[r];

    // O += P @ V  (A = P from LDS, B = V^T rows from LDS)
    const unsigned short* pp = &pb[wid][lm][0];
    v16bf a = frag16(pp + g * 16, pp + g * 16 + 8);
#pragma unroll
    for (int n = 0; n < 8; ++n) {
      const unsigned short* vp = &vt[n * 16 + lm][g * 16];
      v16bf b = frag16(vp, vp + 8);
      o[n] = __builtin_amdgcn_wmma_f32_16x16x32_bf16(false, a, false, b, (short)0, o[n], false, false);
    }
  }

#pragma unroll
  for (int n = 0; n < 8; ++n)
#pragma unroll
    for (int r = 0; r < 8; ++r) {
      int row = qrow + g * 8 + r;
      int col = h * HD + n * 16 + lm;
      O[(size_t)row * DD + col] = f2bf(o[n][r] / lrow[r]);
    }
}

// ======================= launcher =======================
extern "C" void kernel_launch(void* const* d_in, const int* in_sizes, int n_in,
                              void* d_out, int out_size, void* d_ws, size_t ws_size,
                              hipStream_t stream) {
  const float* img   = (const float*)d_in[0];
  const float* txt   = (const float*)d_in[1];
  const float* temb  = (const float*)d_in[2];
  const float* in1w  = (const float*)d_in[3];
  const float* tn1w  = (const float*)d_in[4];
  const float* in2w  = (const float*)d_in[5];
  const float* tn2w  = (const float*)d_in[6];
  const float* qnw   = (const float*)d_in[7];
  const float* knw   = (const float*)d_in[8];
  const float* Wqkv  = (const float*)d_in[9];
  const float* bqkv  = (const float*)d_in[10];
  const float* Wproj = (const float*)d_in[11];
  const float* iguW  = (const float*)d_in[12];
  const float* idnW  = (const float*)d_in[13];
  const float* tguW  = (const float*)d_in[14];
  const float* tdnW  = (const float*)d_in[15];
  const float* imodW = (const float*)d_in[16];
  const float* imodB = (const float*)d_in[17];
  const float* tmodW = (const float*)d_in[18];
  const float* tmodB = (const float*)d_in[19];
  float* out = (float*)d_out;

  float* wsf = (float*)d_ws;
  size_t off = 0;
  float* ts   = wsf + off; off += 4096;
  float* mimg = wsf + off; off += MODN;
  float* mtxt = wsf + off; off += MODN;
  unsigned short* hb    = (unsigned short*)(wsf + off); off += (size_t)LL * DD / 2;
  float*          qkvf  = wsf + off;                    off += (size_t)LL * M3;
  unsigned short* Qb    = (unsigned short*)(wsf + off); off += (size_t)LL * DD / 2;
  unsigned short* Kb    = (unsigned short*)(wsf + off); off += (size_t)LL * DD / 2;
  unsigned short* Vb    = (unsigned short*)(wsf + off); off += (size_t)LL * DD / 2;
  unsigned short* attnb = (unsigned short*)(wsf + off); off += (size_t)LL * DD / 2;
  float*          proj  = wsf + off;                    off += (size_t)LL * DD;
  float*          gu    = wsf + off;                    off += (size_t)LI * GUW;
  unsigned short* actb  = (unsigned short*)(wsf + off); off += (size_t)LI * MMH / 2;
  float*          mlp   = wsf + off;                    off += (size_t)LI * DD;
  unsigned short* wt    = (unsigned short*)(wsf + off); off += (size_t)DD * GUW / 2; // max W^T
  (void)ws_size; (void)n_in; (void)in_sizes; (void)out_size;

  // 1) modulation vectors
  silu_vec_k<<<12, 256, 0, stream>>>(temb, ts, DD);
  modvec_k<<<MODN / 256, 256, 0, stream>>>(ts, imodW, imodB, mimg);
  modvec_k<<<MODN / 256, 256, 0, stream>>>(ts, tmodW, tmodB, mtxt);

  // 2) norm1 + modulate -> joint h1 bf16 (txt rows first)
  norm_mod1_k<<<LL, 256, 0, stream>>>(img, txt, in1w, tn1w, mimg, mtxt, hb);

  // 3) QKV: transpose-convert weight, then WMMA GEMM (+bias) -> fp32
  convtr_k<<<dim3(M3 / 64, DD / 64), 256, 0, stream>>>(Wqkv, wt, DD, M3);
  gemm_bf16_k<<<dim3(M3 / 128, LL / 128), 256, 0, stream>>>(hb, wt, bqkv, qkvf, M3, DD);

  // 4) q/k RMS norm + bf16 pack
  qk_norm_pack_k<<<dim3(LL, NH), 32, 0, stream>>>(qkvf, qnw, knw, Qb, Kb, Vb);

  // 5) attention -> bf16
  attn_flash_k<<<dim3(LL / 128, NH), 256, 0, stream>>>(Qb, Kb, Vb, attnb);

  // 6) output projection
  convtr_k<<<dim3(DD / 64, DD / 64), 256, 0, stream>>>(Wproj, wt, DD, DD);
  gemm_bf16_k<<<dim3(DD / 128, LL / 128), 256, 0, stream>>>(attnb, wt, nullptr, proj, DD, DD);

  // 7) gated residual -> d_out (img rows first)
  resid_attn_k<<<(LL * DD) / 256, 256, 0, stream>>>(img, txt, proj, mimg, mtxt, out);

  // 8) norm2 + modulate -> h2 bf16
  norm_mod2_k<<<LL, 256, 0, stream>>>(out, in2w, tn2w, mimg, mtxt, hb);

  // 9) img MLP
  convtr_k<<<dim3(GUW / 64, DD / 64), 256, 0, stream>>>(iguW, wt, DD, GUW);
  gemm_bf16_k<<<dim3(GUW / 128, LI / 128), 256, 0, stream>>>(hb, wt, nullptr, gu, GUW, DD);
  silu_gate_k<<<(LI * MMH) / 256, 256, 0, stream>>>(gu, actb, LI * MMH);
  convtr_k<<<dim3(DD / 64, MMH / 64), 256, 0, stream>>>(idnW, wt, MMH, DD);
  gemm_bf16_k<<<dim3(DD / 128, LI / 128), 256, 0, stream>>>(actb, wt, nullptr, mlp, DD, MMH);
  resid_mlp_k<<<(LI * DD) / 256, 256, 0, stream>>>(mlp, mimg, out, LI * DD);

  // 10) txt MLP (h2 txt rows start at LI)
  convtr_k<<<dim3(GUW / 64, DD / 64), 256, 0, stream>>>(tguW, wt, DD, GUW);
  gemm_bf16_k<<<dim3(GUW / 128, LT / 128), 256, 0, stream>>>(hb + (size_t)LI * DD, wt, nullptr, gu, GUW, DD);
  silu_gate_k<<<(LT * MMH) / 256, 256, 0, stream>>>(gu, actb, LT * MMH);
  convtr_k<<<dim3(DD / 64, MMH / 64), 256, 0, stream>>>(tdnW, wt, MMH, DD);
  gemm_bf16_k<<<dim3(DD / 128, LT / 128), 256, 0, stream>>>(actb, wt, nullptr, mlp, DD, MMH);
  resid_mlp_k<<<(LT * DD) / 256, 256, 0, stream>>>(mlp, mtxt, out + (size_t)LI * DD, LT * DD);
}
